// MyNet_33775622816290
// MI455X (gfx1250) — matrix operand order repaired and checked by reading, not compiled, not run on hardware
//
#include <hip/hip_runtime.h>
#include <hip/hip_bf16.h>

typedef __attribute__((ext_vector_type(16))) __bf16 bf16x16;
typedef __attribute__((ext_vector_type(8)))  float  floatx8;
typedef unsigned int u32x4 __attribute__((ext_vector_type(4)));
typedef int          i32x4 __attribute__((ext_vector_type(4)));
typedef int          i32x8 __attribute__((ext_vector_type(8)));

struct __attribute__((may_alias, aligned(16))) V4 { unsigned int x, y, z, w; };
struct __attribute__((may_alias, aligned(16))) F4 { float x, y, z, w; };
struct U32x8 { V4 lo, hi; };
typedef float          __attribute__((may_alias)) float_a;
typedef unsigned int   __attribute__((may_alias)) uint_a;
typedef unsigned short __attribute__((may_alias)) ushort_a;
typedef __bf16         __attribute__((may_alias)) bf16_a;

#define WAVES 4
#define TPW   4        // tiles (16 rows each) per wave, double-buffered TDM pipeline
// Per-wave scratch: 2 buffers x 1600 dwords (6400 B each).
//   xs : f32, stride 100 dwords (400 B rows: 16B aligned, conflict-free), 16 rows
//   h1 : bf16, stride 72 hw, 16 rows -> hw [0,1152)     (reuses xs, xs dead)
//   h2 : bf16, stride 40 hw, 16 rows -> hw [1152,1792)
//   h3 : f32,  stride 36,    16 rows -> dword [0,576)   (reuses h1 region)
#define BUF_DW     1600
#define XSF_STRIDE 100
#define H1_STRIDE  72
#define H2_OFF_HW  1152
#define H2_STRIDE  40
#define H3_STRIDE  36

#if __has_builtin(__builtin_amdgcn_tensor_load_to_lds)
#define USE_TDM 1
#else
#define USE_TDM 0
#endif

static __device__ __forceinline__ bf16x16 ld_frag_split(const ushort_a* p0,
                                                        const ushort_a* p1) {
    U32x8 r;
    r.lo = *(const V4*)p0;
    r.hi = *(const V4*)p1;
    return __builtin_bit_cast(bf16x16, r);
}

// gelu: 5 VALU + hardware v_tanh_f32; trans-op hazard covered by s_delay_alu.
static __device__ __forceinline__ float fast_gelu(float x) {
    float x2 = x * x;
    float p  = __builtin_fmaf(0.035677408136300125f, x2, 0.7978845608028654f);
    float u  = x * p;
    float t;
    asm volatile("v_tanh_f32 %0, %1\n\ts_delay_alu instid0(TRANS32_DEP_1)"
                 : "=v"(t) : "v"(u));
    float hx = 0.5f * x;
    return __builtin_fmaf(hx, t, hx);
}

#if USE_TDM
// Issue one TDM: 16 rows x 73 f32, iterate mode, LDS row stride 100 dwords.
static __device__ __forceinline__ void tdm_load_tile(const float* xt, unsigned ldsOff) {
    unsigned long long ga = (unsigned long long)(const void*)xt;
    u32x4 g0;
    g0[0] = 1u;                                                  // count=1
    g0[1] = ldsOff;                                              // lds_addr (bytes)
    g0[2] = (unsigned)ga;                                        // global_addr[31:0]
    g0[3] = (unsigned)((ga >> 32) & 0x1FFFFFFu) | (2u << 30);    // addr[56:32] | type=2
    i32x8 g1;
    g1[0] = (int)((2u << 16) | (1u << 19));   // data_size=4B, iterate_enable
    g1[1] = (int)(73u << 16);                 // tensor_dim0=73 (low 16 @ [31:16])
    g1[2] = (int)(16u << 16);                 // tensor_dim0 hi | tensor_dim1=16 lo
    g1[3] = (int)(73u << 16);                 // tensor_dim1 hi | tile_dim0=73
    g1[4] = 1;                                // tile_dim1=1, tile_dim2=0
    g1[5] = 73;                               // tensor_dim0_stride
    g1[6] = 0;
    g1[7] = 0;
    i32x4 g2;
    g2[0] = 0;
    g2[1] = XSF_STRIDE;                       // lds_addr_increment (elements)
    g2[2] = 73;                               // global_addr_increment (elements)
    g2[3] = (15 << 16);                       // iterate_count = 15 (16 rows)
    i32x4 g3 = {0, 0, 0, 0};
#if defined(__clang_major__) && __clang_major__ >= 23
    i32x8 gz = {0, 0, 0, 0, 0, 0, 0, 0};
    __builtin_amdgcn_tensor_load_to_lds(g0, g1, g2, g3, gz, 0);
#else
    __builtin_amdgcn_tensor_load_to_lds(g0, g1, g2, g3, 0);
#endif
}
#endif

// ---- one-block prep kernel: pack all weight B-fragments (bf16, lane order) once.
// B layout: lane -> N = lane&15, K = (lane>>4)*8 + t (t<8), 16 + (lane>>4)*8 + t-8 (t>=8) within a 32-K chunk.
// Fragments: [0,12) W1 (K padded 73->96, gamma folded), [12,16) W2, [16,18) Wres.
__global__ __launch_bounds__(256) void pack_weights_kernel(
    const float* __restrict__ W1, const float* __restrict__ gamma,
    const float* __restrict__ W2, const float* __restrict__ Wres,
    unsigned short* __restrict__ wpack)
{
    for (int idx = threadIdx.x; idx < 18 * 512; idx += 256) {
        int f = idx >> 9, rem = idx & 511;
        int ln = rem >> 4, t = rem & 15;
        int n_l = ln & 15;
        int khalf = ((ln >> 4) << 3) + (t < 8 ? t : t + 8);
        float v = 0.0f;
        if (f < 12) {
            int nt = f / 3, kc = f - nt * 3;
            int k = kc * 32 + khalf;
            if (k < 73) {
                float w = W1[(nt * 16 + n_l) * 73 + k];
                if (k >= 3 && k < 23) w *= gamma[k];
                v = w;
            }
        } else if (f < 16) {
            int f2 = f - 12, nt = f2 >> 1, kc = f2 & 1;
            v = W2[(nt * 16 + n_l) * 64 + kc * 32 + khalf];
        } else {
            v = Wres[((f - 16) * 16 + n_l) * 32 + khalf];
        }
        ((bf16_a*)wpack)[idx] = (__bf16)v;
    }
}

__global__ __launch_bounds__(128) void mlp_wmma_kernel(
    const float* __restrict__ x,    const float* __restrict__ alpha,
    const float* __restrict__ beta, const unsigned short* __restrict__ wpack,
    const float* __restrict__ b1,   const float* __restrict__ b2,
    const float* __restrict__ bres, const float* __restrict__ W6,
    const float* __restrict__ b6,   const float* __restrict__ gscal,
    float* __restrict__ out, int nRows, int nTiles)
{
    __shared__ unsigned int xbuf[WAVES][2 * BUF_DW];

    const int tid  = threadIdx.x;
    const int lane = tid & 31;
    const int wv   = tid >> 5;
    const int hi   = lane >> 4;
    const int nl   = lane & 15;
    uint_a* bufbase = (uint_a*)&xbuf[wv][0];
    const unsigned ldsOff0 = (unsigned)(unsigned long)(const void*)bufbase;

    const long tbase = ((long)blockIdx.x * WAVES + wv) * TPW;

    // ---- per-wave setup (amortized over TPW tiles) ----------------------------
    // all 18 weight B-fragments -> registers (lane-coalesced, L2-hot)
    bf16x16 wf[18];
    #pragma unroll
    for (int f = 0; f < 18; ++f) {
        const ushort_a* p = (const ushort_a*)wpack + f * 512 + lane * 16;
        U32x8 r;
        r.lo = *(const V4*)p;
        r.hi = *(const V4*)(p + 8);
        wf[f] = __builtin_bit_cast(bf16x16, r);
    }
    // per-lane biases (lane-varying -> VGPRs)
    float bia1[4], bia2[2], bia3[2];
    #pragma unroll
    for (int nt = 0; nt < 4; ++nt) bia1[nt] = b1[nt * 16 + nl];
    #pragma unroll
    for (int nt = 0; nt < 2; ++nt) { bia2[nt] = b2[nt * 16 + nl]; bia3[nt] = bres[nt * 16 + nl]; }
    // uniform w6 -> SGPRs; scalar (b6 + g)
    float w6r[32];
    #pragma unroll
    for (int n = 0; n < 32; ++n) w6r[n] = W6[n];
    const float c0 = b6[0] + gscal[0];

#if USE_TDM
    {   // prime the pipeline: tile 0 of this wave -> buffer 0
        long t0 = tbase < nTiles ? tbase : 0;
        tdm_load_tile(x + t0 * 1168, ldsOff0);
    }
#endif

    #pragma unroll 1
    for (int i = 0; i < TPW; ++i) {
        long tile = tbase + i;
        const bool valid = tile < nTiles;
        if (!valid) tile = 0;
        const long R0 = tile * 16;

        uint_a*   xb  = bufbase + (i & 1) * BUF_DW;
        float_a*  xsf = (float_a*)xb;
        ushort_a* wbh = (ushort_a*)xb;

#if USE_TDM
        if (i + 1 < TPW) {
            long tn = tbase + i + 1;
            if (tn >= nTiles) tn = 0;
            // WAR: prior tile's LDS reads must finish before DMA reuses its buffer
            asm volatile("s_wait_dscnt 0x0" ::: "memory");
            tdm_load_tile(x + tn * 1168, ldsOff0 + ((i + 1) & 1) * (BUF_DW * 4));
            asm volatile("" ::: "memory");
            __builtin_amdgcn_s_wait_tensorcnt(1);   // older DMA (this buffer) done
        } else {
            __builtin_amdgcn_s_wait_tensorcnt(0);
        }
        asm volatile("" ::: "memory");
#else
        {   // fallback: manual coalesced copy (f32), same LDS layout
            const long base  = R0 * 73;
            const long total = (long)nRows * 73;
            for (int i4 = lane; i4 < 292; i4 += 32) {
                float vals[4];
                long g0i = base + (long)i4 * 4;
                if (g0i + 3 < total) {
                    F4 v = *(const F4*)(x + g0i);
                    vals[0] = v.x; vals[1] = v.y; vals[2] = v.z; vals[3] = v.w;
                } else {
                    for (int j = 0; j < 4; ++j) vals[j] = (g0i + j < total) ? x[g0i + j] : 0.0f;
                }
                #pragma unroll
                for (int j = 0; j < 4; ++j) {
                    int idx = i4 * 4 + j;
                    int r = idx / 73, c = idx - r * 73;
                    xsf[r * XSF_STRIDE + c] = vals[j];
                }
            }
        }
#endif

        // ---- embedding fixup: lanes 0..15 -> col0/alpha, lanes 16..31 -> col1/beta
        {
            float_a* slot = xsf + nl * XSF_STRIDE + hi;
            float f = *slot;
            int iv = (int)f;
            if (hi == 0) { iv = iv < 0 ? 0 : (iv > 999  ? 999  : iv); f = alpha[iv]; }
            else         { iv = iv < 0 ? 0 : (iv > 9999 ? 9999 : iv); f = beta[iv];  }
            *slot = f;
        }

        // ---- layer 1: [16x96] x [96x64]; A via hw f32->bf16 cvt; K>=73 masked in regs
        bf16x16 a1[3];
        #pragma unroll
        for (int kc = 0; kc < 2; ++kc) {
            const float_a* p = xsf + nl * XSF_STRIDE + kc * 32 + hi * 8;
            F4 q0 = *(const F4*)p,        q1 = *(const F4*)(p + 4);
            F4 q2 = *(const F4*)(p + 16), q3 = *(const F4*)(p + 20);
            bf16x16 a;
            a[0]  = (__bf16)q0.x; a[1]  = (__bf16)q0.y; a[2]  = (__bf16)q0.z; a[3]  = (__bf16)q0.w;
            a[4]  = (__bf16)q1.x; a[5]  = (__bf16)q1.y; a[6]  = (__bf16)q1.z; a[7]  = (__bf16)q1.w;
            a[8]  = (__bf16)q2.x; a[9]  = (__bf16)q2.y; a[10] = (__bf16)q2.z; a[11] = (__bf16)q2.w;
            a[12] = (__bf16)q3.x; a[13] = (__bf16)q3.y; a[14] = (__bf16)q3.z; a[15] = (__bf16)q3.w;
            a1[kc] = a;
        }
        {   // chunk 2: hi=0 -> K=64..71 valid; hi=1 -> K=72 valid; rest is pad
            const float_a* p = xsf + nl * XSF_STRIDE + 64 + hi * 8;
            F4 q0 = *(const F4*)p, q1 = *(const F4*)(p + 4);
            if (hi) { q0.y = 0.f; q0.z = 0.f; q0.w = 0.f;
                      q1.x = 0.f; q1.y = 0.f; q1.z = 0.f; q1.w = 0.f; }
            bf16x16 a;
            a[0] = (__bf16)q0.x; a[1] = (__bf16)q0.y; a[2] = (__bf16)q0.z; a[3] = (__bf16)q0.w;
            a[4] = (__bf16)q1.x; a[5] = (__bf16)q1.y; a[6] = (__bf16)q1.z; a[7] = (__bf16)q1.w;
            #pragma unroll
            for (int t = 8; t < 16; ++t) a[t] = (__bf16)0.0f;   // K=80..95 pad
            a1[2] = a;
        }
        floatx8 acc1[4];
        #pragma unroll
        for (int nt = 0; nt < 4; ++nt) {
            floatx8 c = {};
            #pragma unroll
            for (int kc = 0; kc < 3; ++kc)
                c = __builtin_amdgcn_wmma_f32_16x16x32_bf16(false, a1[kc], false,
                                                            wf[nt * 3 + kc],
                                                            (short)0, c, false, false);
            acc1[nt] = c;
        }
        #pragma unroll
        for (int nt = 0; nt < 4; ++nt) {         // bias + gelu -> h1 bf16 (reuse xs)
            #pragma unroll
            for (int v = 0; v < 8; ++v) {
                float h = fast_gelu(acc1[nt][v] + bia1[nt]);
                ((bf16_a*)wbh)[(v + hi * 8) * H1_STRIDE + nt * 16 + nl] = (__bf16)h;
            }
        }

        // ---- layer 2: [16x64] x [64x32]; keep post-gelu f32 for residual (4 WMMA)
        bf16x16 a2[2];
        #pragma unroll
        for (int kc = 0; kc < 2; ++kc) {
            const ushort_a* p = wbh + nl * H1_STRIDE + kc * 32 + hi * 8;
            a2[kc] = ld_frag_split(p, p + 16);
        }
        floatx8 h2a[2];
        #pragma unroll
        for (int nt = 0; nt < 2; ++nt) {
            floatx8 c = {};
            #pragma unroll
            for (int kc = 0; kc < 2; ++kc)
                c = __builtin_amdgcn_wmma_f32_16x16x32_bf16(false, a2[kc], false,
                                                            wf[12 + nt * 2 + kc],
                                                            (short)0, c, false, false);
            floatx8 hh;
            #pragma unroll
            for (int v = 0; v < 8; ++v) hh[v] = fast_gelu(c[v] + bia2[nt]);
            h2a[nt] = hh;
            #pragma unroll
            for (int v = 0; v < 8; ++v)
                ((bf16_a*)wbh)[H2_OFF_HW + (v + hi * 8) * H2_STRIDE + nt * 16 + nl] = (__bf16)hh[v];
        }

        // ---- layer 3: gelu(Wres.h2 + bres + h2); residual = register add (2 WMMA)
        const ushort_a* p3 = wbh + H2_OFF_HW + nl * H2_STRIDE + hi * 8;
        bf16x16 a3 = ld_frag_split(p3, p3 + 16);
        float_a* h3 = (float_a*)xb;              // reuse h1 region as f32 [16][36]
        #pragma unroll
        for (int nt = 0; nt < 2; ++nt) {
            floatx8 c = {};
            c = __builtin_amdgcn_wmma_f32_16x16x32_bf16(false, a3, false, wf[16 + nt],
                                                        (short)0, c, false, false);
            #pragma unroll
            for (int v = 0; v < 8; ++v) {
                float h = fast_gelu(c[v] + bia3[nt] + h2a[nt][v]);
                h3[(v + hi * 8) * H3_STRIDE + nt * 16 + nl] = h;
            }
        }

        // ---- layer 4: out = h3 . w6 + (b6 + g); lanes 0..15, b128 LDS reads, w6 in SGPRs
        if (valid && lane < 16) {
            long rg = R0 + lane;
            if (rg < nRows) {
                float s = c0;
                const float_a* hp = h3 + lane * H3_STRIDE;
                #pragma unroll
                for (int q = 0; q < 8; ++q) {
                    F4 hv = *(const F4*)(hp + q * 4);
                    s = __builtin_fmaf(hv.x, w6r[q * 4 + 0], s);
                    s = __builtin_fmaf(hv.y, w6r[q * 4 + 1], s);
                    s = __builtin_fmaf(hv.z, w6r[q * 4 + 2], s);
                    s = __builtin_fmaf(hv.w, w6r[q * 4 + 3], s);
                }
                out[rg] = s;
            }
        }
    }
}

extern "C" void kernel_launch(void* const* d_in, const int* in_sizes, int n_in,
                              void* d_out, int out_size, void* d_ws, size_t ws_size,
                              hipStream_t stream) {
    const float* x     = (const float*)d_in[0];
    const float* alpha = (const float*)d_in[1];
    const float* beta  = (const float*)d_in[2];
    const float* gamma = (const float*)d_in[3];
    const float* g     = (const float*)d_in[4];
    const float* W1    = (const float*)d_in[5];
    const float* b1    = (const float*)d_in[6];
    const float* W2    = (const float*)d_in[7];
    const float* b2    = (const float*)d_in[8];
    const float* Wres  = (const float*)d_in[9];
    const float* bres  = (const float*)d_in[10];
    const float* W6    = (const float*)d_in[11];
    const float* b6    = (const float*)d_in[12];
    float* out = (float*)d_out;
    unsigned short* wpack = (unsigned short*)d_ws;   // 18*512*2 = 18432 B of scratch

    int nRows  = in_sizes[0] / 73;
    int nTiles = (nRows + 15) / 16;
    int tilesPerBlock = WAVES * TPW;
    int blocks = (nTiles + tilesPerBlock - 1) / tilesPerBlock;

    pack_weights_kernel<<<1, 256, 0, stream>>>(W1, gamma, W2, Wres, wpack);
    mlp_wmma_kernel<<<blocks, 128, 0, stream>>>(x, alpha, beta, wpack,
                                                b1, b2, bres, W6, b6, g,
                                                out, nRows, nTiles);
}
